// Qwen2DyMemDecoderLayer_6614249636544
// MI455X (gfx1250) — matrix operand order repaired and checked by reading, not compiled, not run on hardware
//
#include <hip/hip_runtime.h>
#include <hip/hip_bf16.h>

typedef __bf16 bf16_t;
typedef __attribute__((ext_vector_type(16))) __bf16 v16bf;
typedef __attribute__((ext_vector_type(8)))  __bf16 v8bf;
typedef __attribute__((ext_vector_type(4)))  __bf16 v4bf;
typedef __attribute__((ext_vector_type(8)))  float  v8f;
typedef __attribute__((ext_vector_type(4)))  float  v4f;

#define WMMA_BF16(a,b,c) __builtin_amdgcn_wmma_f32_16x16x32_bf16(false,(a),false,(b),(short)0,(c),false,false)

constexpr int T_   = 4096;
constexpr int HID_ = 2048;
constexpr int NH_  = 16;
constexpr int NKV_ = 4;
constexpr int HD_  = 128;
constexpr int IM_  = 8192;

// ---- CDNA5 async global->LDS copy (16B per lane), tracked by ASYNCcnt ----
__device__ inline void async_copy_b128(uint32_t lds_byte_addr, const void* gaddr) {
  asm volatile("global_load_async_to_lds_b128 %0, %1, off"
               :: "v"(lds_byte_addr), "v"(gaddr) : "memory");
}
__device__ inline void wait_async0() {
  asm volatile("s_wait_asynccnt 0x0" ::: "memory");
}
__device__ inline uint32_t lds_addr_of(const void* p) {
  return (uint32_t)(uintptr_t)p;   // LDS aperture: low 32 bits are the LDS byte offset
}

// ---------------- fragment helpers (CDNA5 wave32 WMMA VGPR layouts) ----------------
// A 16x32 bf16: lane holds M=lane&15; element e: K = kbase + e (16 contiguous), kbase = k0 + (lane>=16?16:0)
__device__ inline v16bf a_frag_from_lds(const bf16_t* row_base /* points at kbase */) {
  v8bf lo = *(const v8bf*)(row_base);
  v8bf hi = *(const v8bf*)(row_base + 8);
  v16bf a;
#pragma unroll
  for (int i = 0; i < 8; ++i) { a[i] = lo[i]; a[8 + i] = hi[i]; }
  return a;
}
// B 32x16 bf16: lane holds N=lane&15; element e: K = kbase + e (16 contiguous), kbase = k0 + (lane>=16?16:0)
__device__ inline v16bf b_frag_from_lds(const bf16_t* base) {
  v8bf lo = *(const v8bf*)(base);
  v8bf hi = *(const v8bf*)(base + 8);
  v16bf b;
#pragma unroll
  for (int i = 0; i < 8; ++i) { b[i] = lo[i]; b[8 + i] = hi[i]; }
  return b;
}
__device__ inline v16bf a_frag_from_f32(const float* __restrict__ row, int kbase) {
  v4f c0 = *(const v4f*)(row + kbase);
  v4f c1 = *(const v4f*)(row + kbase + 4);
  v4f c2 = *(const v4f*)(row + kbase + 8);
  v4f c3 = *(const v4f*)(row + kbase + 12);
  v16bf a;
#pragma unroll
  for (int i = 0; i < 4; ++i) {
    a[i] = (bf16_t)c0[i]; a[4 + i] = (bf16_t)c1[i];
    a[8 + i] = (bf16_t)c2[i]; a[12 + i] = (bf16_t)c3[i];
  }
  return a;
}

// ---------------- fp32 -> bf16 conversion (weights / activations) ----------------
__global__ void cvt_bf16_kernel(const float* __restrict__ in, bf16_t* __restrict__ out, int n) {
  int i = (blockIdx.x * blockDim.x + threadIdx.x) * 4;
  const int stride = gridDim.x * blockDim.x * 4;
  for (; i < n; i += stride) {
    v4f x = *(const v4f*)(in + i);
    v4bf y;
#pragma unroll
    for (int j = 0; j < 4; ++j) y[j] = (bf16_t)x[j];
    *(v4bf*)(out + i) = y;
  }
}

// ---------------- RMSNorm (templated output type) ----------------
template <typename OutT>
__global__ void rmsnorm_kernel(const float* __restrict__ x, const float* __restrict__ w,
                               OutT* __restrict__ out, int ncols) {
  __shared__ float red[256];
  const int row = blockIdx.x;
  const float* xr = x + (size_t)row * ncols;
  float s = 0.f;
  for (int c = threadIdx.x; c < ncols; c += blockDim.x) { float v = xr[c]; s += v * v; }
  red[threadIdx.x] = s;
  __syncthreads();
  for (int off = 128; off > 0; off >>= 1) {
    if ((int)threadIdx.x < off) red[threadIdx.x] += red[threadIdx.x + off];
    __syncthreads();
  }
  const float inv = rsqrtf(red[0] / (float)ncols + 1e-6f);
  OutT* orow = out + (size_t)row * ncols;
  for (int c = threadIdx.x; c < ncols; c += blockDim.x) orow[c] = (OutT)(xr[c] * inv * w[c]);
}

// ---------------- RoPE (scale folded in; for Q scale = 1/sqrt(HD)) ----------------
__global__ void rope_kernel(float* __restrict__ x, const float* __restrict__ cosb,
                            const float* __restrict__ sinb, int stride, float scale) {
  __shared__ float sh[HD_];
  const int t = blockIdx.x, h = blockIdx.y, d = threadIdx.x;
  float* p = x + (size_t)t * stride + h * HD_;
  sh[d] = p[d];
  __syncthreads();
  const float c = cosb[t * HD_ + d];
  const float s = sinb[t * HD_ + d];
  const float rot = (d < HD_ / 2) ? -sh[d + HD_ / 2] : sh[d - HD_ / 2];
  p[d] = (sh[d] * c + rot * s) * scale;
}

// ---------------- residual add ----------------
__global__ void add_kernel(const float* __restrict__ a, const float* __restrict__ b,
                           float* __restrict__ o, int n) {
  int i = blockIdx.x * blockDim.x + threadIdx.x;
  const int stride = gridDim.x * blockDim.x;
  for (; i < n; i += stride) o[i] = a[i] + b[i];
}

// ---------------- tiled GEMM: C[M,N] = A[M,K](bf16) @ W[N,K]^T(bf16) (+bias) ----------------
// Block = 256 threads (8 waves). Tile 128x128, BK=32. Double-buffered async LDS copies.
__global__ __launch_bounds__(256)
void tiled_gemm_kernel(const bf16_t* __restrict__ A, const bf16_t* __restrict__ W,
                       const float* __restrict__ bias, float* __restrict__ C,
                       int M, int N, int K) {
  __shared__ bf16_t ldsA[2][128 * 32];
  __shared__ bf16_t ldsB[2][128 * 32];
  const int t = threadIdx.x;
  const int lane = t & 31, wave = t >> 5;
  const int l15 = lane & 15, half = lane >> 4;
  const int tileM = blockIdx.x * 128;
  const int tileN = blockIdx.y * 128;

  auto issue = [&](int buf, int k0) {
#pragma unroll
    for (int r = 0; r < 2; ++r) {
      const int idx = t + r * 256;                // 16B chunk id; 4 chunks per 32-elem row
      const int row = idx >> 2, cc = (idx & 3) * 8;
      async_copy_b128(lds_addr_of(&ldsA[buf][idx * 8]),
                      A + (size_t)(tileM + row) * K + k0 + cc);
      async_copy_b128(lds_addr_of(&ldsB[buf][idx * 8]),
                      W + (size_t)(tileN + row) * K + k0 + cc);
    }
  };

  v8f acc[8];
#pragma unroll
  for (int j = 0; j < 8; ++j) acc[j] = (v8f){0.f, 0.f, 0.f, 0.f, 0.f, 0.f, 0.f, 0.f};

  issue(0, 0);
  wait_async0();
  __syncthreads();

  int buf = 0;
  for (int k0 = 0; k0 < K; k0 += 32) {
    if (k0 + 32 < K) issue(buf ^ 1, k0 + 32);
    const v16bf af = a_frag_from_lds(&ldsA[buf][(wave * 16 + l15) * 32 + half * 16]);
#pragma unroll
    for (int j = 0; j < 8; ++j) {
      const v16bf bfrag = b_frag_from_lds(&ldsB[buf][(j * 16 + l15) * 32 + half * 16]);
      acc[j] = WMMA_BF16(af, bfrag, acc[j]);
    }
    wait_async0();
    __syncthreads();
    buf ^= 1;
  }

  const int row0 = tileM + wave * 16 + half * 8;
#pragma unroll
  for (int j = 0; j < 8; ++j) {
    const int col = tileN + j * 16 + l15;
    const float bv = bias ? bias[col] : 0.f;
#pragma unroll
    for (int r = 0; r < 8; ++r)
      C[(size_t)(row0 + r) * N + col] = acc[j][r] + bv;
  }
}

// ---------------- fused gate/up tiled GEMM + SiLU*up -> bf16 ----------------
// Tile 128(M) x 64(N per weight), BK=32, double-buffered async copies.
__global__ __launch_bounds__(256)
void gateup_silu_kernel(const bf16_t* __restrict__ A, const bf16_t* __restrict__ Wg,
                        const bf16_t* __restrict__ Wu, bf16_t* __restrict__ act,
                        int M, int N, int K) {
  __shared__ bf16_t ldsA[2][128 * 32];
  __shared__ bf16_t ldsG[2][64 * 32];
  __shared__ bf16_t ldsU[2][64 * 32];
  const int t = threadIdx.x;
  const int lane = t & 31, wave = t >> 5;
  const int l15 = lane & 15, half = lane >> 4;
  const int tileM = blockIdx.x * 128;
  const int tileN = blockIdx.y * 64;

  auto issue = [&](int buf, int k0) {
#pragma unroll
    for (int r = 0; r < 2; ++r) {
      const int idx = t + r * 256;
      const int row = idx >> 2, cc = (idx & 3) * 8;
      async_copy_b128(lds_addr_of(&ldsA[buf][idx * 8]),
                      A + (size_t)(tileM + row) * K + k0 + cc);
    }
    {
      const int row = t >> 2, cc = (t & 3) * 8;
      async_copy_b128(lds_addr_of(&ldsG[buf][t * 8]),
                      Wg + (size_t)(tileN + row) * K + k0 + cc);
      async_copy_b128(lds_addr_of(&ldsU[buf][t * 8]),
                      Wu + (size_t)(tileN + row) * K + k0 + cc);
    }
  };

  v8f accg[4], accu[4];
#pragma unroll
  for (int j = 0; j < 4; ++j) {
    accg[j] = (v8f){0.f, 0.f, 0.f, 0.f, 0.f, 0.f, 0.f, 0.f};
    accu[j] = (v8f){0.f, 0.f, 0.f, 0.f, 0.f, 0.f, 0.f, 0.f};
  }

  issue(0, 0);
  wait_async0();
  __syncthreads();

  int buf = 0;
  for (int k0 = 0; k0 < K; k0 += 32) {
    if (k0 + 32 < K) issue(buf ^ 1, k0 + 32);
    const v16bf af = a_frag_from_lds(&ldsA[buf][(wave * 16 + l15) * 32 + half * 16]);
#pragma unroll
    for (int j = 0; j < 4; ++j) {
      const v16bf bg = b_frag_from_lds(&ldsG[buf][(j * 16 + l15) * 32 + half * 16]);
      accg[j] = WMMA_BF16(af, bg, accg[j]);
      const v16bf bu = b_frag_from_lds(&ldsU[buf][(j * 16 + l15) * 32 + half * 16]);
      accu[j] = WMMA_BF16(af, bu, accu[j]);
    }
    wait_async0();
    __syncthreads();
    buf ^= 1;
  }

  const int row0 = tileM + wave * 16 + half * 8;
#pragma unroll
  for (int j = 0; j < 4; ++j) {
    const int col = tileN + j * 16 + l15;
#pragma unroll
    for (int r = 0; r < 8; ++r) {
      const float g = accg[j][r];
      const float silu = g / (1.f + __expf(-g));
      act[(size_t)(row0 + r) * N + col] = (bf16_t)(silu * accu[j][r]);
    }
  }
}

// ---------------- flash attention: one wave per (head, 16-query block) ----------------
__global__ void flash_attn_kernel(const float* __restrict__ qb, const float* __restrict__ kb,
                                  const float* __restrict__ vb, bf16_t* __restrict__ ob, int T) {
  const int lane = threadIdx.x;
  const int l15 = lane & 15, half = lane >> 4;
  const int h = blockIdx.y;
  const int qt = blockIdx.x * 16;
  const int kv = h >> 2;               // NH/NKV = 4
  const int qstride = NH_ * HD_;       // 2048
  const int kstride = NKV_ * HD_;      // 512
  __shared__ bf16_t ldsP[16 * 32];

  v16bf qf[4];
  const float* qrow = qb + (size_t)(qt + l15) * qstride + h * HD_;
#pragma unroll
  for (int c = 0; c < 4; ++c) qf[c] = a_frag_from_f32(qrow, c * 32 + half * 16);

  float mrow[8], lrow[8];
  v8f oacc[8];
#pragma unroll
  for (int r = 0; r < 8; ++r) {
    mrow[r] = -3.0e38f;
    lrow[r] = 0.f;
    oacc[r] = (v8f){0.f, 0.f, 0.f, 0.f, 0.f, 0.f, 0.f, 0.f};
  }

  for (int kt = 0; kt < T; kt += 32) {
    v8f s0 = {0.f, 0.f, 0.f, 0.f, 0.f, 0.f, 0.f, 0.f};
    v8f s1 = {0.f, 0.f, 0.f, 0.f, 0.f, 0.f, 0.f, 0.f};
#pragma unroll
    for (int c = 0; c < 4; ++c) {
      const int kd = c * 32 + half * 16;
      const float* k0r = kb + (size_t)(kt + l15) * kstride + kv * HD_ + kd;
      const float* k1r = kb + (size_t)(kt + 16 + l15) * kstride + kv * HD_ + kd;
      v16bf b0, b1;
#pragma unroll
      for (int j = 0; j < 4; ++j) {
        v4f c0 = *(const v4f*)(k0r + 4 * j);
        v4f c1 = *(const v4f*)(k1r + 4 * j);
#pragma unroll
        for (int i = 0; i < 4; ++i) { b0[4 * j + i] = (bf16_t)c0[i]; b1[4 * j + i] = (bf16_t)c1[i]; }
      }
      s0 = WMMA_BF16(qf[c], b0, s0);
      s1 = WMMA_BF16(qf[c], b1, s1);
    }

#pragma unroll
    for (int r = 0; r < 8; ++r) {
      float a = s0[r], b = s1[r];
      float mx = fmaxf(a, b);
#pragma unroll
      for (int off = 1; off < 16; off <<= 1) mx = fmaxf(mx, __shfl_xor(mx, off, 32));
      const float mnew = fmaxf(mrow[r], mx);
      const float p0 = __expf(a - mnew);
      const float p1 = __expf(b - mnew);
      float sum = p0 + p1;
#pragma unroll
      for (int off = 1; off < 16; off <<= 1) sum += __shfl_xor(sum, off, 32);
      const float sc = __expf(mrow[r] - mnew);
      lrow[r] = lrow[r] * sc + sum;
      mrow[r] = mnew;
#pragma unroll
      for (int n = 0; n < 8; ++n) oacc[n][r] *= sc;
      const int prow = r + half * 8;
      ldsP[prow * 32 + l15] = (bf16_t)p0;
      ldsP[prow * 32 + 16 + l15] = (bf16_t)p1;
    }
    __syncthreads();

    v16bf pf;
    {
      const int kb0 = half * 16;
#pragma unroll
      for (int e = 0; e < 8; ++e) {
        pf[e] = ldsP[l15 * 32 + kb0 + e];
        pf[8 + e] = ldsP[l15 * 32 + kb0 + 8 + e];
      }
    }
    __syncthreads();

#pragma unroll
    for (int n = 0; n < 8; ++n) {
      const float* vcol = vb + (size_t)(kt + half * 16) * kstride + kv * HD_ + n * 16 + l15;
      v16bf vf;
#pragma unroll
      for (int e = 0; e < 16; ++e) vf[e] = (bf16_t)vcol[(size_t)e * kstride];
      oacc[n] = WMMA_BF16(pf, vf, oacc[n]);
    }
  }

#pragma unroll
  for (int n = 0; n < 8; ++n) {
#pragma unroll
    for (int r = 0; r < 8; ++r) {
      ob[(size_t)(qt + r + half * 8) * qstride + h * HD_ + n * 16 + l15] =
          (bf16_t)(oacc[n][r] / lrow[r]);
    }
  }
}

// ---------------- orchestration ----------------
extern "C" void kernel_launch(void* const* d_in, const int* in_sizes, int n_in,
                              void* d_out, int out_size, void* d_ws, size_t ws_size,
                              hipStream_t stream) {
  (void)in_sizes; (void)n_in; (void)out_size; (void)ws_size;
  const float* hidden = (const float*)d_in[0];
  const float* cosb   = (const float*)d_in[1];
  const float* sinb   = (const float*)d_in[2];
  const float* w_q    = (const float*)d_in[3];
  const float* b_q    = (const float*)d_in[4];
  const float* w_k    = (const float*)d_in[5];
  const float* b_k    = (const float*)d_in[6];
  const float* w_v    = (const float*)d_in[7];
  const float* b_v    = (const float*)d_in[8];
  const float* w_o    = (const float*)d_in[9];
  const float* w_gate = (const float*)d_in[10];
  const float* w_up   = (const float*)d_in[11];
  const float* w_down = (const float*)d_in[12];
  const float* w_ln1  = (const float*)d_in[13];
  const float* w_ln2  = (const float*)d_in[14];
  float* out = (float*)d_out;

  char* ws = (char*)d_ws;
  const size_t MB = 1024 * 1024;
  bf16_t* wq_b = (bf16_t*)(ws + 0 * MB);     //  8MB (4M)
  bf16_t* wk_b = (bf16_t*)(ws + 8 * MB);     //  2MB (1M)
  bf16_t* wv_b = (bf16_t*)(ws + 10 * MB);    //  2MB (1M)
  bf16_t* wo_b = (bf16_t*)(ws + 12 * MB);    //  8MB (4M)
  bf16_t* wg_b = (bf16_t*)(ws + 20 * MB);    // 32MB (16M)
  bf16_t* wu_b = (bf16_t*)(ws + 52 * MB);    // 32MB
  bf16_t* wd_b = (bf16_t*)(ws + 84 * MB);    // 32MB -> weights end 116MB
  bf16_t* xn_b = (bf16_t*)(ws + 116 * MB);   // 16MB (also h2 later)
  float*  q    = (float*)(ws + 132 * MB);    // 32MB
  float*  k    = (float*)(ws + 164 * MB);    //  8MB
  float*  v    = (float*)(ws + 172 * MB);    //  8MB
  bf16_t* o_b  = (bf16_t*)(ws + 180 * MB);   // 16MB
  float*  attn = (float*)(ws + 196 * MB);    // 32MB (also mlp out later)
  float*  h    = (float*)(ws + 228 * MB);    // 32MB -> total 260MB
  bf16_t* act_b = (bf16_t*)(ws + 132 * MB);  // 64MB, reuses q/k/v/o_b (dead by then)
  float*  mlp  = attn;                       // reuse

  // 0) weights -> bf16 (cheap one-shot; makes every GEMM operand bf16)
  cvt_bf16_kernel<<<dim3(2048), dim3(256), 0, stream>>>(w_q, wq_b, HID_ * HID_);
  cvt_bf16_kernel<<<dim3(2048), dim3(256), 0, stream>>>(w_k, wk_b, NKV_ * HD_ * HID_);
  cvt_bf16_kernel<<<dim3(2048), dim3(256), 0, stream>>>(w_v, wv_b, NKV_ * HD_ * HID_);
  cvt_bf16_kernel<<<dim3(2048), dim3(256), 0, stream>>>(w_o, wo_b, HID_ * HID_);
  cvt_bf16_kernel<<<dim3(2048), dim3(256), 0, stream>>>(w_gate, wg_b, IM_ * HID_);
  cvt_bf16_kernel<<<dim3(2048), dim3(256), 0, stream>>>(w_up, wu_b, IM_ * HID_);
  cvt_bf16_kernel<<<dim3(2048), dim3(256), 0, stream>>>(w_down, wd_b, HID_ * IM_);

  // 1) x = rmsnorm(hidden, w_ln1) -> bf16
  rmsnorm_kernel<bf16_t><<<dim3(T_), dim3(256), 0, stream>>>(hidden, w_ln1, xn_b, HID_);

  // 2) QKV projections (tiled WMMA + async LDS)
  tiled_gemm_kernel<<<dim3(T_ / 128, HID_ / 128), dim3(256), 0, stream>>>(
      xn_b, wq_b, b_q, q, T_, HID_, HID_);
  tiled_gemm_kernel<<<dim3(T_ / 128, (NKV_ * HD_) / 128), dim3(256), 0, stream>>>(
      xn_b, wk_b, b_k, k, T_, NKV_ * HD_, HID_);
  tiled_gemm_kernel<<<dim3(T_ / 128, (NKV_ * HD_) / 128), dim3(256), 0, stream>>>(
      xn_b, wv_b, b_v, v, T_, NKV_ * HD_, HID_);

  // 3) RoPE (1/sqrt(HD) folded into Q)
  rope_kernel<<<dim3(T_, NH_), dim3(HD_), 0, stream>>>(q, cosb, sinb, NH_ * HD_, 0.08838834764831845f);
  rope_kernel<<<dim3(T_, NKV_), dim3(HD_), 0, stream>>>(k, cosb, sinb, NKV_ * HD_, 1.0f);

  // 4) attention (flash, WMMA) -> bf16
  flash_attn_kernel<<<dim3(T_ / 16, NH_), dim3(32), 0, stream>>>(q, k, v, o_b, T_);

  // 5) O projection
  tiled_gemm_kernel<<<dim3(T_ / 128, HID_ / 128), dim3(256), 0, stream>>>(
      o_b, wo_b, nullptr, attn, T_, HID_, HID_);

  // 6) h = hidden + attn
  add_kernel<<<dim3(4096), dim3(256), 0, stream>>>(hidden, attn, h, T_ * HID_);

  // 7) h2 = rmsnorm(h, w_ln2) -> bf16 (reuse xn_b)
  rmsnorm_kernel<bf16_t><<<dim3(T_), dim3(256), 0, stream>>>(h, w_ln2, xn_b, HID_);

  // 8) act = silu(h2 @ Wg^T) * (h2 @ Wu^T) -> bf16
  gateup_silu_kernel<<<dim3(T_ / 128, IM_ / 64), dim3(256), 0, stream>>>(
      xn_b, wg_b, wu_b, act_b, T_, IM_, HID_);

  // 9) mlp = act @ Wd^T
  tiled_gemm_kernel<<<dim3(T_ / 128, HID_ / 128), dim3(256), 0, stream>>>(
      act_b, wd_b, nullptr, mlp, T_, HID_, IM_);

  // 10) out = h + mlp
  add_kernel<<<dim3(4096), dim3(256), 0, stream>>>(h, mlp, out, T_ * HID_);
}
